// MultiHeadLatentAttention_48490180772350
// MI455X (gfx1250) — matrix operand order repaired and checked by reading, not compiled, not run on hardware
//
#include <hip/hip_runtime.h>
#include <hip/hip_bf16.h>

typedef __bf16 v16bf __attribute__((ext_vector_type(16)));
typedef __bf16 v8bf  __attribute__((ext_vector_type(8)));
typedef float  v8f   __attribute__((ext_vector_type(8)));
typedef unsigned int uint32x4 __attribute__((ext_vector_type(4)));
typedef int int32x4 __attribute__((ext_vector_type(4)));
typedef int int32x8 __attribute__((ext_vector_type(8)));

#define HIDDEN 1024
#define NH     16
#define RDIM   64
#define SEQ    2048
#define BATCH  2
#define NTOK   (BATCH*SEQ)   // 4096
#define LDSLD  72            // 64 elems + 16B TDM pad -> conflict-free stride

// ---------------------------------------------------------------- helpers ---

static __device__ __forceinline__ v16bf cat8(v8bf lo, v8bf hi) {
  v16bf r;
#pragma unroll
  for (int i = 0; i < 8; ++i) { r[i] = lo[i]; r[i + 8] = hi[i]; }
  return r;
}

// A fragment (16x32 bf16): lane m(+16*hi), elems 0-7 -> K=hi*8+0..7,
// elems 8-15 -> K=hi*8+16..23 (two 16-byte loads).
static __device__ __forceinline__ v16bf load_a(const __bf16* p, int ld) {
  const int lane = threadIdx.x & 31;
  const int m = lane & 15, hi = lane >> 4;
  const __bf16* q = p + (size_t)m * ld + hi * 8;
  return cat8(*(const v8bf*)q, *(const v8bf*)(q + 16));
}

// B fragment (32x16 bf16) with B = W^T, W row-major [N,K]:
// lane n(+16*hi): elems 0-7 -> K = hi*8 + 0..7, elems 8-15 -> K = hi*8 + 16..23.
static __device__ __forceinline__ v16bf load_bt(const __bf16* w, int ld) {
  const int lane = threadIdx.x & 31;
  const int n = lane & 15, hi = lane >> 4;
  const __bf16* q = w + (size_t)n * ld + hi * 8;
  return cat8(*(const v8bf*)q, *(const v8bf*)(q + 16));
}

static __device__ __forceinline__ v8f wmma_bf16(v16bf a, v16bf b, v8f c) {
  return __builtin_amdgcn_wmma_f32_16x16x32_bf16(false, a, false, b,
                                                 (short)0, c, false, false);
}

// --------------------------------------------------- Tensor Data Mover (D#) -
// 2D tile of bf16: rows x cols, source row-major with stride (elems).
// LDS gets 16B padding after every 128B (row of 64 bf16) -> row stride 144B.

static __device__ __forceinline__ void tdm_load_2d(const void* lds_dst,
                                                   const void* gsrc,
                                                   unsigned rows,
                                                   unsigned cols,
                                                   unsigned stride_elems) {
  const unsigned lds_off = (unsigned)(uintptr_t)lds_dst;
  const unsigned long long ga = (unsigned long long)(uintptr_t)gsrc;
  uint32x4 g0;
  g0[0] = 1u;                                        // count=1, user mode
  g0[1] = lds_off;                                   // lds_addr (bytes)
  g0[2] = (unsigned)(ga & 0xFFFFFFFFu);              // global_addr lo
  g0[3] = (unsigned)((ga >> 32) & 0x01FFFFFFu) | (2u << 30);  // hi | type=2
  int32x8 g1;
  g1[0] = (int)((1u << 16)      // data_size = 1 -> 2-byte elements
              | (1u << 20)      // pad_enable
              | (4u << 22)      // pad_interval = 4 -> every 32 DWORDs (128B)
              | (3u << 25));    // pad_amount   = 3 -> 4 DWORDs (16B)
  g1[1] = (int)((cols & 0xFFFFu) << 16);                       // tensor_dim0 lo
  g1[2] = (int)(((cols >> 16) & 0xFFFFu) | ((rows & 0xFFFFu) << 16)); // d0 hi | d1 lo
  g1[3] = (int)(((rows >> 16) & 0xFFFFu) | ((cols & 0xFFFFu) << 16)); // d1 hi | tile_dim0
  g1[4] = (int)(rows & 0xFFFFu);                     // tile_dim1 | tile_dim2=0
  g1[5] = (int)stride_elems;                         // tensor_dim0_stride lo32
  g1[6] = 0;                                         // stride hi | dim1_stride lo
  g1[7] = 0;
  const int32x4 z4 = {0, 0, 0, 0};                   // 2D: groups 2/3 unused
  const int32x8 z8 = {0, 0, 0, 0, 0, 0, 0, 0};
  __builtin_amdgcn_tensor_load_to_lds(g0, g1, z4, z4, z8, 0);
}

// ------------------------------------------------------------ prep kernels --

__global__ void cvt_bf16_kernel(const float* __restrict__ src,
                                __bf16* __restrict__ dst, int n) {
  int i = blockIdx.x * 256 + threadIdx.x;
  if (i < n) dst[i] = (__bf16)src[i];
}

// Weff[h*64+r, k] = sum_d Wl[r, d] * W[h*64+d, k]   (folds latent projection)
__global__ void build_weff_kernel(const float* __restrict__ W,
                                  const float* __restrict__ Wl,
                                  __bf16* __restrict__ dst) {
  int idx = blockIdx.x * 256 + threadIdx.x;
  int k = idx & (HIDDEN - 1);
  int o = idx >> 10;
  int h = o >> 6, r = o & 63;
  float acc = 0.f;
#pragma unroll 4
  for (int d = 0; d < 64; ++d)
    acc += Wl[r * 64 + d] * W[(size_t)(h * 64 + d) * HIDDEN + k];
  dst[idx] = (__bf16)acc;
}

__global__ void build_beff_kernel(const float* __restrict__ b,
                                  const float* __restrict__ Wl,
                                  const float* __restrict__ bl,
                                  float* __restrict__ dst) {
  int o = blockIdx.x * 256 + threadIdx.x;
  int h = o >> 6, r = o & 63;
  float acc = bl[r];
#pragma unroll 4
  for (int d = 0; d < 64; ++d) acc += Wl[r * 64 + d] * b[h * 64 + d];
  dst[o] = acc;
}

// -------------------------------------------- fused QKV + latent projection -
// C = Xbf[4096,1024] @ Weff^T + beff, TDM-staged LDS double buffering.
// Block = 4 waves; wave owns M=32 (2 A frags, 8 accumulators) x N=64.
// z=0 -> Ql [B,nh,S,64], z=1 -> Kl, z=2 -> Vl transposed [B,nh,64,S]

__global__ __launch_bounds__(128)
void qkv_gemm_kernel(const __bf16* __restrict__ X,
                     const __bf16* __restrict__ Wqe, const __bf16* __restrict__ Wke,
                     const __bf16* __restrict__ Wve,
                     const float* __restrict__ bqe, const float* __restrict__ bke,
                     const float* __restrict__ bve,
                     __bf16* __restrict__ Ql, __bf16* __restrict__ Kl,
                     __bf16* __restrict__ VlT) {
  __shared__ __bf16 As[2][128][LDSLD];   // 36 KB
  __shared__ __bf16 Bs[2][64][LDSLD];    // 18 KB

  const int z = blockIdx.z;
  const __bf16* W   = (z == 0) ? Wqe : ((z == 1) ? Wke : Wve);
  const float* bias = (z == 0) ? bqe : ((z == 1) ? bke : bve);

  const int wave = threadIdx.x >> 5;             // 0..3
  const int m0 = blockIdx.x * 128;               // token rows of block
  const int h  = blockIdx.y;                     // head (64-wide N tile)
  const int n0 = h * 64;

  if (wave == 0) {
    tdm_load_2d(&As[0][0][0], X + (size_t)m0 * HIDDEN, 128, 64, HIDDEN);
    tdm_load_2d(&Bs[0][0][0], W + (size_t)n0 * HIDDEN, 64, 64, HIDDEN);
  }

  v8f acc[2][4] = {};
  for (int kt = 0; kt < HIDDEN / 64; ++kt) {
    const int cur = kt & 1;
    __syncthreads();                             // next buffer free
    if (wave == 0) {
      if (kt + 1 < HIDDEN / 64) {
        tdm_load_2d(&As[cur ^ 1][0][0], X + (size_t)m0 * HIDDEN + (kt + 1) * 64,
                    128, 64, HIDDEN);
        tdm_load_2d(&Bs[cur ^ 1][0][0], W + (size_t)n0 * HIDDEN + (kt + 1) * 64,
                    64, 64, HIDDEN);
        __builtin_amdgcn_s_wait_tensorcnt(2);    // current pair landed
      } else {
        __builtin_amdgcn_s_wait_tensorcnt(0);
      }
    }
    __syncthreads();                             // current buffer ready
#pragma unroll
    for (int ks = 0; ks < 2; ++ks) {
      // batch all fragment loads, then 8 independent WMMAs
      const v16bf a0 = load_a(&As[cur][wave * 32 + 0][0]  + ks * 32, LDSLD);
      const v16bf a1 = load_a(&As[cur][wave * 32 + 16][0] + ks * 32, LDSLD);
      const v16bf b0 = load_bt(&Bs[cur][0][0]  + ks * 32, LDSLD);
      const v16bf b1 = load_bt(&Bs[cur][16][0] + ks * 32, LDSLD);
      const v16bf b2 = load_bt(&Bs[cur][32][0] + ks * 32, LDSLD);
      const v16bf b3 = load_bt(&Bs[cur][48][0] + ks * 32, LDSLD);
      acc[0][0] = wmma_bf16(a0, b0, acc[0][0]);
      acc[0][1] = wmma_bf16(a0, b1, acc[0][1]);
      acc[0][2] = wmma_bf16(a0, b2, acc[0][2]);
      acc[0][3] = wmma_bf16(a0, b3, acc[0][3]);
      acc[1][0] = wmma_bf16(a1, b0, acc[1][0]);
      acc[1][1] = wmma_bf16(a1, b1, acc[1][1]);
      acc[1][2] = wmma_bf16(a1, b2, acc[1][2]);
      acc[1][3] = wmma_bf16(a1, b3, acc[1][3]);
    }
  }

  const int lane = threadIdx.x & 31;
  const int n = lane & 15, hi = lane >> 4;
#pragma unroll
  for (int mi = 0; mi < 2; ++mi)
#pragma unroll
    for (int nf = 0; nf < 4; ++nf) {
      const int rl = nf * 16 + n;                // latent index within head
      const float bb = bias[n0 + rl];
#pragma unroll
      for (int r = 0; r < 8; ++r) {
        const int tok = m0 + wave * 32 + mi * 16 + r + 8 * hi;
        const int b_ = tok >> 11;
        const int s  = tok & (SEQ - 1);
        const __bf16 v = (__bf16)(acc[mi][nf][r] + bb);
        if (z == 0)
          Ql[((size_t)(b_ * NH + h) * SEQ + s) * RDIM + rl] = v;
        else if (z == 1)
          Kl[((size_t)(b_ * NH + h) * SEQ + s) * RDIM + rl] = v;
        else
          VlT[((size_t)(b_ * NH + h) * RDIM + rl) * SEQ + s] = v;
      }
    }
}

// ----------------------------------------------------------- flash attention -
// Block = 4 waves x 16 queries; K/V tiles (64 keys) TDM-staged + double
// buffered in LDS, shared by all waves; online softmax; P transposed via LDS.

__global__ __launch_bounds__(128)
void flash_attn_kernel(const __bf16* __restrict__ Ql,
                       const __bf16* __restrict__ Kl,
                       const __bf16* __restrict__ VlT,
                       __bf16* __restrict__ attnB) {
  __shared__ __bf16 Ks[2][64][LDSLD];            // 18 KB
  __shared__ __bf16 Vs[2][64][LDSLD];            // 18 KB
  __shared__ __bf16 Plds[4][16][LDSLD];          // 9 KB (per-wave P tiles)

  const int wave = threadIdx.x >> 5;
  const int lane = threadIdx.x & 31;
  const int n = lane & 15, hi = lane >> 4;

  const int bh = blockIdx.y;                     // b*NH + h
  const int q0 = blockIdx.x * 64 + wave * 16;

  const __bf16* Qb = Ql  + (size_t)bh * SEQ * RDIM;
  const __bf16* Kb = Kl  + (size_t)bh * SEQ * RDIM;
  const __bf16* Vb = VlT + (size_t)bh * RDIM * SEQ;

  const v16bf aq0 = load_a(Qb + (size_t)q0 * RDIM + 0,  RDIM);
  const v16bf aq1 = load_a(Qb + (size_t)q0 * RDIM + 32, RDIM);

  if (wave == 0) {
    tdm_load_2d(&Ks[0][0][0], Kb, 64, 64, RDIM);   // keys x r
    tdm_load_2d(&Vs[0][0][0], Vb, 64, 64, SEQ);    // r x keys (VlT rows)
  }

  v8f of[4] = {};
  float mrow[8], lrow[8];
#pragma unroll
  for (int r = 0; r < 8; ++r) { mrow[r] = -3.0e38f; lrow[r] = 0.f; }
  const float scale = 0.125f;                    // 1/sqrt(64)

  for (int kt = 0; kt < SEQ / 64; ++kt) {
    const int cur = kt & 1;
    __syncthreads();                             // next buffer free
    if (wave == 0) {
      if (kt + 1 < SEQ / 64) {
        tdm_load_2d(&Ks[cur ^ 1][0][0], Kb + (size_t)(kt + 1) * 64 * RDIM,
                    64, 64, RDIM);
        tdm_load_2d(&Vs[cur ^ 1][0][0], Vb + (kt + 1) * 64, 64, 64, SEQ);
        __builtin_amdgcn_s_wait_tensorcnt(2);
      } else {
        __builtin_amdgcn_s_wait_tensorcnt(0);
      }
    }
    __syncthreads();                             // current K/V ready

    // S = Ql(16x64) @ Kl^T(64x64): batch all 8 B frags, then 8 WMMAs
    v16bf bk[8];
#pragma unroll
    for (int nf = 0; nf < 4; ++nf) {
      bk[2 * nf + 0] = load_bt(&Ks[cur][nf * 16][0] + 0,  LDSLD);
      bk[2 * nf + 1] = load_bt(&Ks[cur][nf * 16][0] + 32, LDSLD);
    }
    v8f sf[4] = {};
#pragma unroll
    for (int nf = 0; nf < 4; ++nf) sf[nf] = wmma_bf16(aq0, bk[2 * nf], sf[nf]);
#pragma unroll
    for (int nf = 0; nf < 4; ++nf) sf[nf] = wmma_bf16(aq1, bk[2 * nf + 1], sf[nf]);
#pragma unroll
    for (int nf = 0; nf < 4; ++nf)
#pragma unroll
      for (int r = 0; r < 8; ++r) sf[nf][r] *= scale;

    // online softmax; row stats per lane for rows r + 8*hi
    float alpha[8];
#pragma unroll
    for (int r = 0; r < 8; ++r) {
      float v = fmaxf(fmaxf(sf[0][r], sf[1][r]), fmaxf(sf[2][r], sf[3][r]));
#pragma unroll
      for (int off = 8; off; off >>= 1) v = fmaxf(v, __shfl_xor(v, off, 32));
      const float mnew = fmaxf(mrow[r], v);
      alpha[r] = __expf(mrow[r] - mnew);
      mrow[r] = mnew;
    }

    float rsum[8];
#pragma unroll
    for (int r = 0; r < 8; ++r) rsum[r] = 0.f;
#pragma unroll
    for (int nf = 0; nf < 4; ++nf)
#pragma unroll
      for (int r = 0; r < 8; ++r) {
        const float p = __expf(sf[nf][r] - mrow[r]);
        rsum[r] += p;
        Plds[wave][r + 8 * hi][nf * 16 + n] = (__bf16)p;   // C-layout scatter
      }
#pragma unroll
    for (int r = 0; r < 8; ++r) {
      float v = rsum[r];
#pragma unroll
      for (int off = 8; off; off >>= 1) v += __shfl_xor(v, off, 32);
      lrow[r] = lrow[r] * alpha[r] + v;
    }
#pragma unroll
    for (int nf = 0; nf < 4; ++nf)
#pragma unroll
      for (int r = 0; r < 8; ++r) of[nf][r] *= alpha[r];

    __syncthreads();                             // P visible for A-layout read
    const v16bf ap0 = load_a(&Plds[wave][0][0],  LDSLD);
    const v16bf ap1 = load_a(&Plds[wave][0][32], LDSLD);

    // O += P(16x64) @ Vl(64x64): batch all 8 B frags, then 8 WMMAs
    v16bf bv[8];
#pragma unroll
    for (int nf = 0; nf < 4; ++nf) {
      bv[2 * nf + 0] = load_bt(&Vs[cur][nf * 16][0] + 0,  LDSLD);
      bv[2 * nf + 1] = load_bt(&Vs[cur][nf * 16][0] + 32, LDSLD);
    }
#pragma unroll
    for (int nf = 0; nf < 4; ++nf) of[nf] = wmma_bf16(ap0, bv[2 * nf], of[nf]);
#pragma unroll
    for (int nf = 0; nf < 4; ++nf) of[nf] = wmma_bf16(ap1, bv[2 * nf + 1], of[nf]);
  }

  const int b_ = bh >> 4;
  const int h  = bh & 15;
#pragma unroll
  for (int nf = 0; nf < 4; ++nf)
#pragma unroll
    for (int r = 0; r < 8; ++r) {
      const int tok = b_ * SEQ + q0 + r + 8 * hi;
      const float v = of[nf][r] / lrow[r];
      attnB[(size_t)tok * HIDDEN + h * 64 + nf * 16 + n] = (__bf16)v;
    }
}

// ------------------------------------------------------------- output GEMM --

__global__ __launch_bounds__(128)
void out_gemm_kernel(const __bf16* __restrict__ A, const __bf16* __restrict__ Wo,
                     const float* __restrict__ bo, float* __restrict__ out) {
  __shared__ __bf16 As[2][128][LDSLD];
  __shared__ __bf16 Bs[2][64][LDSLD];

  const int wave = threadIdx.x >> 5;             // 0..3
  const int m0 = blockIdx.x * 128;
  const int n0 = blockIdx.y * 64;

  if (wave == 0) {
    tdm_load_2d(&As[0][0][0], A + (size_t)m0 * HIDDEN, 128, 64, HIDDEN);
    tdm_load_2d(&Bs[0][0][0], Wo + (size_t)n0 * HIDDEN, 64, 64, HIDDEN);
  }

  v8f acc[2][4] = {};
  for (int kt = 0; kt < HIDDEN / 64; ++kt) {
    const int cur = kt & 1;
    __syncthreads();
    if (wave == 0) {
      if (kt + 1 < HIDDEN / 64) {
        tdm_load_2d(&As[cur ^ 1][0][0], A + (size_t)m0 * HIDDEN + (kt + 1) * 64,
                    128, 64, HIDDEN);
        tdm_load_2d(&Bs[cur ^ 1][0][0], Wo + (size_t)n0 * HIDDEN + (kt + 1) * 64,
                    64, 64, HIDDEN);
        __builtin_amdgcn_s_wait_tensorcnt(2);
      } else {
        __builtin_amdgcn_s_wait_tensorcnt(0);
      }
    }
    __syncthreads();
#pragma unroll
    for (int ks = 0; ks < 2; ++ks) {
      const v16bf a0 = load_a(&As[cur][wave * 32 + 0][0]  + ks * 32, LDSLD);
      const v16bf a1 = load_a(&As[cur][wave * 32 + 16][0] + ks * 32, LDSLD);
      const v16bf b0 = load_bt(&Bs[cur][0][0]  + ks * 32, LDSLD);
      const v16bf b1 = load_bt(&Bs[cur][16][0] + ks * 32, LDSLD);
      const v16bf b2 = load_bt(&Bs[cur][32][0] + ks * 32, LDSLD);
      const v16bf b3 = load_bt(&Bs[cur][48][0] + ks * 32, LDSLD);
      acc[0][0] = wmma_bf16(a0, b0, acc[0][0]);
      acc[0][1] = wmma_bf16(a0, b1, acc[0][1]);
      acc[0][2] = wmma_bf16(a0, b2, acc[0][2]);
      acc[0][3] = wmma_bf16(a0, b3, acc[0][3]);
      acc[1][0] = wmma_bf16(a1, b0, acc[1][0]);
      acc[1][1] = wmma_bf16(a1, b1, acc[1][1]);
      acc[1][2] = wmma_bf16(a1, b2, acc[1][2]);
      acc[1][3] = wmma_bf16(a1, b3, acc[1][3]);
    }
  }

  const int lane = threadIdx.x & 31;
  const int n = lane & 15, hi = lane >> 4;
#pragma unroll
  for (int mi = 0; mi < 2; ++mi)
#pragma unroll
    for (int nf = 0; nf < 4; ++nf) {
      const float bb = bo[n0 + nf * 16 + n];
#pragma unroll
      for (int r = 0; r < 8; ++r) {
        const int tok = m0 + wave * 32 + mi * 16 + r + 8 * hi;
        out[(size_t)tok * HIDDEN + n0 + nf * 16 + n] = acc[mi][nf][r] + bb;
      }
    }
}

// ---------------------------------------------------------------- launcher --

extern "C" void kernel_launch(void* const* d_in, const int* in_sizes, int n_in,
                              void* d_out, int out_size, void* d_ws, size_t ws_size,
                              hipStream_t stream) {
  const float* X  = (const float*)d_in[0];
  const float* Wq = (const float*)d_in[1];
  const float* bq = (const float*)d_in[2];
  const float* Wk = (const float*)d_in[3];
  const float* bk = (const float*)d_in[4];
  const float* Wv = (const float*)d_in[5];
  const float* bv = (const float*)d_in[6];
  const float* Wl = (const float*)d_in[7];
  const float* bl = (const float*)d_in[8];
  const float* Wo = (const float*)d_in[9];
  const float* bo = (const float*)d_in[10];
  float* out = (float*)d_out;

  char* ws = (char*)d_ws;
  size_t off = 0;
  auto alloc = [&](size_t bytes) -> char* {
    char* p = ws + off;
    off = (off + bytes + 255) & ~(size_t)255;
    return p;
  };
  __bf16* Xbf  = (__bf16*)alloc((size_t)NTOK * HIDDEN * 2);
  __bf16* WoBf = (__bf16*)alloc((size_t)HIDDEN * HIDDEN * 2);
  __bf16* Wqe  = (__bf16*)alloc((size_t)HIDDEN * HIDDEN * 2);
  __bf16* Wke  = (__bf16*)alloc((size_t)HIDDEN * HIDDEN * 2);
  __bf16* Wve  = (__bf16*)alloc((size_t)HIDDEN * HIDDEN * 2);
  float*  bqe  = (float*)alloc(HIDDEN * 4);
  float*  bke  = (float*)alloc(HIDDEN * 4);
  float*  bve  = (float*)alloc(HIDDEN * 4);
  __bf16* Qlb  = (__bf16*)alloc((size_t)BATCH * NH * SEQ * RDIM * 2);
  __bf16* Klb  = (__bf16*)alloc((size_t)BATCH * NH * SEQ * RDIM * 2);
  __bf16* VlTb = (__bf16*)alloc((size_t)BATCH * NH * RDIM * SEQ * 2);
  __bf16* attnB = (__bf16*)alloc((size_t)NTOK * HIDDEN * 2);

  {
    int nx = NTOK * HIDDEN;
    cvt_bf16_kernel<<<(nx + 255) / 256, 256, 0, stream>>>(X, Xbf, nx);
    int nw = HIDDEN * HIDDEN;
    cvt_bf16_kernel<<<(nw + 255) / 256, 256, 0, stream>>>(Wo, WoBf, nw);
    build_weff_kernel<<<(HIDDEN * HIDDEN) / 256, 256, 0, stream>>>(Wq, Wl, Wqe);
    build_weff_kernel<<<(HIDDEN * HIDDEN) / 256, 256, 0, stream>>>(Wk, Wl, Wke);
    build_weff_kernel<<<(HIDDEN * HIDDEN) / 256, 256, 0, stream>>>(Wv, Wl, Wve);
    build_beff_kernel<<<HIDDEN / 256, 256, 0, stream>>>(bq, Wl, bl, bqe);
    build_beff_kernel<<<HIDDEN / 256, 256, 0, stream>>>(bk, Wl, bl, bke);
    build_beff_kernel<<<HIDDEN / 256, 256, 0, stream>>>(bv, Wl, bl, bve);
  }
  {
    dim3 g(NTOK / 128, NH, 3);
    qkv_gemm_kernel<<<g, 128, 0, stream>>>(Xbf, Wqe, Wke, Wve, bqe, bke, bve,
                                           Qlb, Klb, VlTb);
  }
  {
    dim3 g(SEQ / 64, BATCH * NH);
    flash_attn_kernel<<<g, 128, 0, stream>>>(Qlb, Klb, VlTb, attnB);
  }
  {
    dim3 g(NTOK / 128, HIDDEN / 64);
    out_gemm_kernel<<<g, 128, 0, stream>>>(attnB, WoBf, bo, out);
  }
}
